// DecoderRNN_75067438399697
// MI455X (gfx1250) — matrix operand order repaired and checked by reading, not compiled, not run on hardware
//
#include <hip/hip_runtime.h>
#include <hip/hip_bf16.h>

typedef __attribute__((ext_vector_type(16))) __bf16 v16bf;
typedef __attribute__((ext_vector_type(8)))  __bf16 v8bf;
typedef __attribute__((ext_vector_type(8)))  float  v8f;

#define H     512
#define OUTC  66
#define NCLS  12
#define KIN   96      // OUT+NCLS (78) padded to 3*32; col 78 carries constant 1.0 (bias row)
#define G4    2048    // 4*H
#define FCN   128     // fc out cols padded (power of two row stride)
#define LEN   64
#define BT    16      // batch rows per workgroup
#define BATCH 512
#define BROW  78      // augmented bias row index

static __device__ __forceinline__ unsigned short f2bf(float f) {
    unsigned int u = __float_as_uint(f);
    unsigned int r = (u + 0x7FFFu + ((u >> 16) & 1u)) >> 16;   // RNE
    return (unsigned short)r;
}

// A-matrix tile (16x32 bf16) from LDS, per-lane layout:
// lane<16:  row=lane,    K = k0..k0+7 , k0+16..k0+23
// lane>=16: row=lane-16, K = k0+8..15, k0+24..31   (caller folds +8 into k0)
static __device__ __forceinline__ v16bf ldsA(const unsigned short* buf, int stride,
                                             int m, int k0) {
    const v8bf* p0 = (const v8bf*)(buf + m * stride + k0);
    const v8bf* p1 = (const v8bf*)(buf + m * stride + k0 + 16);
    v8bf a = *p0, b = *p1;
    return __builtin_shufflevector(a, b, 0,1,2,3,4,5,6,7,8,9,10,11,12,13,14,15);
}

// B-matrix tile (32x16 bf16) from transposed global weights: lane holds column n0+lane%16,
// K = k0+8h..+7, k0+16+8h..+7 (h = lane/16), matching ldsA.
static __device__ __forceinline__ v16bf gldB(const unsigned short* __restrict__ W,
                                             int stride, int k0, int n0, int lane) {
    const __bf16* Wb = (const __bf16*)W;
    int n = n0 + (lane & 15), h = lane >> 4;
    v16bf r;
#pragma unroll
    for (int s = 0; s < 16; ++s)
        r[s] = Wb[(size_t)(k0 + 8 * h + (s & 7) + ((s >> 3) << 4)) * stride + n];
    return r;
}

static __device__ __forceinline__ v8f wmma_bf16(v16bf a, v16bf b, v8f c) {
    return __builtin_amdgcn_wmma_f32_16x16x32_bf16(false, a, false, b, (short)0, c,
                                                   false, false);
}

// store h (C-layout v8f per hidden tile) into LDS as bf16 row-major [16][H]
static __device__ __forceinline__ void store_h(unsigned short* hbuf, const v8f* hv,
                                               int hb, int lmod, int lhalf) {
#pragma unroll
    for (int ht = 0; ht < 4; ++ht) {
        int col = hb + ht * 16 + lmod;
#pragma unroll
        for (int r = 0; r < 8; ++r)
            hbuf[(r + lhalf * 8) * H + col] = f2bf(hv[ht][r]);
    }
}

// ---------- prep: transpose + pad + fp32->bf16 ----------
// W is [N, Ksrc] row-major; Wt is [Kdst, Ndst]; Wt[k][n] = (k<Ksrc && n<N) ? W[n][k] : 0
__global__ void prep_tp(const float* __restrict__ W, unsigned short* __restrict__ Wt,
                        int N, int Ksrc, int Kdst, int Ndst) {
    int idx = blockIdx.x * 256 + threadIdx.x;
    if (idx >= Kdst * Ndst) return;
    int k = idx / Ndst, n = idx % Ndst;
    float v = (k < Ksrc && n < N) ? W[n * Ksrc + k] : 0.f;
    Wt[idx] = f2bf(v);
}

// write bias rows (row 78) into the augmented-K weight matrices (after prep_tp zeroed them)
__global__ void prep_biasrow(const float* __restrict__ b_ih, const float* __restrict__ b_hh,
                             const float* __restrict__ b_inh, const float* __restrict__ b_inc,
                             unsigned short* __restrict__ WihT,
                             unsigned short* __restrict__ WinhT,
                             unsigned short* __restrict__ WincT) {
    int i = blockIdx.x * 256 + threadIdx.x;
    if (i < G4) WihT[BROW * G4 + i] = f2bf(b_ih[i] + b_hh[i]);
    if (i < H) {
        WinhT[BROW * H + i] = f2bf(b_inh[i]);
        WincT[BROW * H + i] = f2bf(b_inc[i]);
    }
}

// ---------- main fused decoder ----------
__global__ __launch_bounds__(256, 1)
void decoder_rnn(const float* __restrict__ inputs, const int* __restrict__ labels,
                 const unsigned short* __restrict__ WihT, const unsigned short* __restrict__ WhhT,
                 const unsigned short* __restrict__ WfcT, const unsigned short* __restrict__ WinhT,
                 const unsigned short* __restrict__ WincT, const float* __restrict__ b_fc,
                 float* __restrict__ out) {
    __shared__ unsigned short frame[BT * KIN];   // [16][96]  bf16 (frame0|onehot|1.0|0)
    __shared__ unsigned short hbuf [BT * H];     // [16][512] bf16

    const int tid   = threadIdx.x;
    const int wave  = tid >> 5;
    const int lane  = tid & 31;
    const int lmod  = lane & 15;
    const int lhalf = lane >> 4;
    const int wgbase = blockIdx.x * BT;
    const int hb     = wave * 64;                // this wave's hidden-column base
    const v8f  vzero = {};                       // lowers to inline SRC2=0 in WMMA

    // gate-column base for local tile t: gate (t>>2) of hidden block (t&3)
    int ncol[16];
#pragma unroll
    for (int t = 0; t < 16; ++t) ncol[t] = (t >> 2) * H + hb + (t & 3) * 16;

    // stage initial frame: [frame0(66) | one_hot(12) | 1.0 | zeros]
    for (int e = tid; e < BT * KIN; e += 256) {
        int m = e / KIN, k = e % KIN;
        float v = 0.f;
        if (k < OUTC)             v = inputs[(wgbase + m) * OUTC + k];
        else if (k < OUTC + NCLS) v = (labels[wgbase + m] == (k - OUTC)) ? 1.f : 0.f;
        else if (k == BROW)       v = 1.f;       // bias row activation
        frame[e] = f2bf(v);
    }
    __syncthreads();

    // ---- initial state GEMMs: h0/c0 = [frame0|1] @ [Winh|b]^T  (bias via row 78) ----
    v8f hv[4], cst[4];
#pragma unroll
    for (int ht = 0; ht < 4; ++ht) { hv[ht] = vzero; cst[ht] = vzero; }
#pragma unroll
    for (int kc = 0; kc < KIN / 32; ++kc) {
        v16bf a = ldsA(frame, KIN, lmod, kc * 32 + lhalf * 8);
#pragma unroll
        for (int ht = 0; ht < 4; ++ht) {
            hv[ht]  = wmma_bf16(a, gldB(WinhT, H, kc * 32, hb + ht * 16, lane), hv[ht]);
            cst[ht] = wmma_bf16(a, gldB(WincT, H, kc * 32, hb + ht * 16, lane), cst[ht]);
        }
    }
    store_h(hbuf, hv, hb, lmod, lhalf);
    __syncthreads();

    // ---- 64 recurrent steps ----
    for (int step = 0; step < LEN; ++step) {
        // gates[16, 2048]: this wave owns 16 tiles (i/f/g/o x 4 hidden tiles).
        // Biases arrive through Wih row 78 * frame[.,78]==1, so acc starts at inline 0.
        v8f acc[16];
#pragma unroll
        for (int t = 0; t < 16; ++t) acc[t] = vzero;

#pragma unroll 1
        for (int kc = 0; kc < KIN / 32; ++kc) {           // input (+bias) part
            v16bf a = ldsA(frame, KIN, lmod, kc * 32 + lhalf * 8);
#pragma unroll
            for (int t = 0; t < 16; ++t)
                acc[t] = wmma_bf16(a, gldB(WihT, G4, kc * 32, ncol[t], lane), acc[t]);
        }
#pragma unroll 1
        for (int kc = 0; kc < H / 32; ++kc) {             // recurrent part
            v16bf a = ldsA(hbuf, H, lmod, kc * 32 + lhalf * 8);
#pragma unroll
            for (int t = 0; t < 16; ++t)
                acc[t] = wmma_bf16(a, gldB(WhhT, G4, kc * 32, ncol[t], lane), acc[t]);
        }
        __syncthreads();   // all hbuf/frame reads done before overwriting hbuf

        // LSTM cell (f32, register-resident c state, torch gate order i,f,g,o)
#pragma unroll
        for (int ht = 0; ht < 4; ++ht) {
#pragma unroll
            for (int r = 0; r < 8; ++r) {
                float iv = 1.f / (1.f + __expf(-acc[ht][r]));
                float fv = 1.f / (1.f + __expf(-acc[4 + ht][r]));
                float gv = tanhf(acc[8 + ht][r]);
                float ov = 1.f / (1.f + __expf(-acc[12 + ht][r]));
                float cn = fv * cst[ht][r] + iv * gv;
                cst[ht][r] = cn;
                hv[ht][r]  = ov * tanhf(cn);
            }
        }
        store_h(hbuf, hv, hb, lmod, lhalf);
        __syncthreads();   // h_new visible to fc waves

        // fc projection: waves 0..4 produce out cols [16w, 16w+16) (wave-uniform branch)
        if (wave < 5) {
            int n0  = wave * 16;
            int col = n0 + lmod;
            float bo = (col < OUTC) ? b_fc[col] : 0.f;
            v8f oc = vzero;
#pragma unroll 4
            for (int kc = 0; kc < H / 32; ++kc) {
                v16bf a = ldsA(hbuf, H, lmod, kc * 32 + lhalf * 8);
                oc = wmma_bf16(a, gldB(WfcT, FCN, kc * 32, n0, lane), oc);
            }
            if (col < OUTC) {
#pragma unroll
                for (int r = 0; r < 8; ++r) {
                    int row = r + lhalf * 8;
                    float ov = oc[r] + bo;      // bias folded at store
                    out[((size_t)(wgbase + row) * LEN + step) * OUTC + col] = ov;
                    frame[row * KIN + col] = f2bf(ov);   // next step's input
                }
            }
        }
        __syncthreads();   // frame ready for next step
    }
}

extern "C" void kernel_launch(void* const* d_in, const int* in_sizes, int n_in,
                              void* d_out, int out_size, void* d_ws, size_t ws_size,
                              hipStream_t stream) {
    const float* inputs = (const float*)d_in[0];
    const int*   labels = (const int*)d_in[1];
    // d_in[2] = length (fixed at 64)
    const float* W_ih  = (const float*)d_in[3];
    const float* W_hh  = (const float*)d_in[4];
    const float* b_ih  = (const float*)d_in[5];
    const float* b_hh  = (const float*)d_in[6];
    const float* W_fc  = (const float*)d_in[7];
    const float* b_fc  = (const float*)d_in[8];
    const float* W_inh = (const float*)d_in[9];
    const float* b_inh = (const float*)d_in[10];
    const float* W_inc = (const float*)d_in[11];
    const float* b_inc = (const float*)d_in[12];

    char* ws = (char*)d_ws;
    unsigned short* WihT  = (unsigned short*)(ws + 0);        // [96][2048]   393216 B
    unsigned short* WhhT  = (unsigned short*)(ws + 393216);   // [512][2048] 2097152 B
    unsigned short* WfcT  = (unsigned short*)(ws + 2490368);  // [512][128]   131072 B
    unsigned short* WinhT = (unsigned short*)(ws + 2621440);  // [96][512]     98304 B
    unsigned short* WincT = (unsigned short*)(ws + 2719744);  // [96][512]     98304 B

    prep_tp<<<(96 * 2048 + 255) / 256, 256, 0, stream>>>(W_ih, WihT, 2048, 78, 96, 2048);
    prep_tp<<<(512 * 2048 + 255) / 256, 256, 0, stream>>>(W_hh, WhhT, 2048, 512, 512, 2048);
    prep_tp<<<(512 * 128 + 255) / 256, 256, 0, stream>>>(W_fc, WfcT, 66, 512, 512, 128);
    prep_tp<<<(96 * 512 + 255) / 256, 256, 0, stream>>>(W_inh, WinhT, 512, 66, 96, 512);
    prep_tp<<<(96 * 512 + 255) / 256, 256, 0, stream>>>(W_inc, WincT, 512, 66, 96, 512);
    prep_biasrow<<<(G4 + 255) / 256, 256, 0, stream>>>(b_ih, b_hh, b_inh, b_inc,
                                                       WihT, WinhT, WincT);

    decoder_rnn<<<BATCH / BT, 256, 0, stream>>>(inputs, labels, WihT, WhhT, WfcT,
                                                WinhT, WincT, b_fc, (float*)d_out);
}
